// ConvNextLayer_30262339568086
// MI455X (gfx1250) — matrix operand order repaired and checked by reading, not compiled, not run on hardware
//
#include <hip/hip_runtime.h>
#include <stdint.h>

#define FEAT 64
#define ATTR 32
#define HID  256
#define LN_EPS 1e-5f

typedef __attribute__((ext_vector_type(16))) __bf16 v16bf;
typedef __attribute__((ext_vector_type(8)))  float  v8f;
typedef __attribute__((ext_vector_type(4)))  float  v4f;
typedef __attribute__((ext_vector_type(4)))  unsigned int v4u;

union FragU {
    v16bf v;
    v4u   q[2];
    unsigned short s[16];
};

__device__ __forceinline__ unsigned short f2bf_raw(float f) {
    union { __bf16 b; unsigned short s; } u;
    u.b = (__bf16)f;                       // native RNE f32->bf16 convert
    return u.s;
}

// ---------------- zero the accumulator (d_out doubles as aggr) --------------
__global__ void zero_kernel(float4* out, int n4) {
    int i = blockIdx.x * blockDim.x + threadIdx.x;
    if (i < n4) out[i] = make_float4(0.f, 0.f, 0.f, 0.f);
}

// ------------- pack weights into bf16 WMMA B-fragment order in ws -----------
// frag F, lane l: 16 halfs at ws[(F*32+l)*16].  F: 0..3 Wk | 4..35 W1 | 36..67 W2
__global__ void prep_weights(const float* __restrict__ Wk,
                             const float* __restrict__ W1,
                             const float* __restrict__ W2,
                             unsigned short* __restrict__ ws) {
    int idx = blockIdx.x * blockDim.x + threadIdx.x;   // frag*32 + lane
    if (idx >= 68 * 32) return;
    int F = idx >> 5, l = idx & 31;
    int n  = l & 15;
    int kb = (l & 16) >> 1;                            // B-layout: +8 for upper half-wave
    unsigned short* d = ws + idx * 16;
    if (F < 4) {                                       // W_kernel [64,32]: B[k][n]=Wk[nb*16+n][k]
        int nb = F;
        for (int h = 0; h < 16; ++h)
            d[h] = f2bf_raw(Wk[(nb * 16 + n) * ATTR + ((h & 7) + ((h & 8) << 1) + kb)]);
    } else if (F < 36) {                               // W1 [256,64]
        int G = F - 4, nb = G >> 1, ks = G & 1;
        for (int h = 0; h < 16; ++h)
            d[h] = f2bf_raw(W1[(nb * 16 + n) * FEAT + (ks * 32 + (h & 7) + ((h & 8) << 1) + kb)]);
    } else {                                           // W2 [64,256]
        int G = F - 36, n2 = G >> 3, ks = G & 7;
        for (int h = 0; h < 16; ++h)
            d[h] = f2bf_raw(W2[(n2 * 16 + n) * HID + (ks * 32 + (h & 7) + ((h & 8) << 1) + kb)]);
    }
}

// ---- edge stage: kernel = attr@Wᵀ (WMMA), msg = kernel*x[src], scatter-add --
__global__ void edge_kernel(const float* __restrict__ x,
                            const float* __restrict__ attr,
                            const int*   __restrict__ src,
                            const int*   __restrict__ dst,
                            const unsigned short* __restrict__ wsfrag,
                            float* aggr, int numTiles, int totalWaves) {
    int l    = threadIdx.x & 31;
    int wave = (blockIdx.x * blockDim.x + threadIdx.x) >> 5;
    int hb   = (l & 16) >> 1;                          // A-frag / D-frag half offset
    int n    = l & 15;

    v16bf B[4];                                        // Wᵀ B-fragments, resident
    #pragma unroll
    for (int nb = 0; nb < 4; ++nb) {
        FragU u;
        const v4u* p = (const v4u*)(wsfrag + (nb * 32 + l) * 16);
        u.q[0] = p[0]; u.q[1] = p[1];
        B[nb] = u.v;
    }

    for (int t = wave; t < numTiles; t += totalWaves) {
        int e0 = t * 16;
        const float* arow = attr + (e0 + (l & 15)) * ATTR;
        __builtin_prefetch(attr + (e0 + 16 * totalWaves) * ATTR, 0, 0);

        // A fragment: two contiguous 32B runs per lane -> 4x global_load_b128
        const v4f* a4 = (const v4f*)arow;
        int cb = hb >> 2;                              // 0 or 2
        v4f c0 = a4[cb], c1 = a4[cb + 1], c2 = a4[cb + 4], c3 = a4[cb + 5];
        float va[16];
        *(v4f*)&va[0]  = c0; *(v4f*)&va[4]  = c1;
        *(v4f*)&va[8]  = c2; *(v4f*)&va[12] = c3;
        FragU a;
        #pragma unroll
        for (int h = 0; h < 16; ++h) a.s[h] = f2bf_raw(va[h]);

        v8f D[4];
        #pragma unroll
        for (int nb = 0; nb < 4; ++nb) {
            v8f c = {0.f,0.f,0.f,0.f,0.f,0.f,0.f,0.f};
            D[nb] = __builtin_amdgcn_wmma_f32_16x16x32_bf16(
                        false, a.v, false, B[nb], (short)0, c, false, false);
        }

        // batched epilogue: all index loads, then all gathers, then all atomics
        int sA[8], dA[8];
        #pragma unroll
        for (int r = 0; r < 8; ++r) {
            int em = e0 + r + hb;
            sA[r] = src[em];
            dA[r] = dst[em];
        }
        float xv[8][4];
        #pragma unroll
        for (int r = 0; r < 8; ++r) {
            const float* xr = x + sA[r] * FEAT + n;
            #pragma unroll
            for (int nb = 0; nb < 4; ++nb) xv[r][nb] = xr[nb * 16];
        }
        #pragma unroll
        for (int r = 0; r < 8; ++r) {
            float* ar = aggr + dA[r] * FEAT + n;
            #pragma unroll
            for (int nb = 0; nb < 4; ++nb)
                atomicAdd(ar + nb * 16, D[nb][r] * xv[r][nb]);
        }
    }
}

// ------- node stage: LN -> MLP(64->256 relu ->64) -> scale + residual -------
#define WPB 4
__global__ void node_kernel(const float* __restrict__ x,
                            const float* aggrin,       // aliases out (d_out)
                            const float* __restrict__ conv_bias,
                            const float* __restrict__ gamma,
                            const float* __restrict__ beta,
                            const float* __restrict__ b1,
                            const float* __restrict__ b2,
                            const float* __restrict__ lscale,
                            const unsigned short* __restrict__ wsfrag,
                            float* out, int numTiles) {
    __shared__ unsigned short lds_ln[WPB][16 * FEAT];
    __shared__ unsigned short lds_h1[WPB][16 * HID];

    int l = threadIdx.x & 31;
    int w = threadIdx.x >> 5;
    int tile = blockIdx.x * WPB + w;
    if (tile >= numTiles) return;

    int node0 = tile * 16;
    int m  = l & 15;
    int hs = l >> 4;                                   // 0: feats 0..31, 1: feats 32..63
    int hb = (l & 16) >> 1;
    int n  = l & 15;

    // ---- layernorm (f32), lane pair (l, l^16) covers one row ----
    const v4f* a4 = (const v4f*)(aggrin + (node0 + m) * FEAT + hs * 32);
    const v4f* cb4 = (const v4f*)(conv_bias + hs * 32);
    float vbuf[32], sum = 0.f, sumsq = 0.f;
    #pragma unroll
    for (int i = 0; i < 8; ++i) {
        v4f v = a4[i] + cb4[i];
        *(v4f*)&vbuf[i * 4] = v;
        #pragma unroll
        for (int j = 0; j < 4; ++j) { sum += v[j]; sumsq += v[j] * v[j]; }
    }
    sum   += __shfl_xor(sum,   16, 32);
    sumsq += __shfl_xor(sumsq, 16, 32);
    float mu  = sum * (1.f / 64.f);
    float var = sumsq * (1.f / 64.f) - mu * mu;
    float rs  = rsqrtf(var + LN_EPS);

    const v4f* g4 = (const v4f*)(gamma + hs * 32);
    const v4f* be4 = (const v4f*)(beta + hs * 32);
    unsigned short lnloc[32];
    #pragma unroll
    for (int i = 0; i < 8; ++i) {
        v4f g = g4[i], be = be4[i];
        #pragma unroll
        for (int j = 0; j < 4; ++j)
            lnloc[i * 4 + j] = f2bf_raw((vbuf[i * 4 + j] - mu) * rs * g[j] + be[j]);
    }
    // 4x ds_store_b128 (16B-aligned)
    v4u* lnrow = (v4u*)&lds_ln[w][m * FEAT + hs * 32];
    #pragma unroll
    for (int c = 0; c < 4; ++c) lnrow[c] = ((v4u*)lnloc)[c];

    // ---- A-fragments of LN output: 2x ds_load_b128 each ----
    v16bf A[2];
    #pragma unroll
    for (int ks = 0; ks < 2; ++ks) {
        FragU u;
        const v4u* lp = (const v4u*)&lds_ln[w][m * FEAT + ks * 32 + hb];
        u.q[0] = lp[0];                                // k = hb .. hb+7
        u.q[1] = lp[2];                                // k = hb+16 .. hb+23
        A[ks] = u.v;
    }

    // ---- GEMM1: [16x64] @ W1ᵀ[64x256], +b1, relu -> LDS h1 (bf16) ----
    #pragma unroll 4
    for (int nb = 0; nb < 16; ++nb) {
        v8f c = {0.f,0.f,0.f,0.f,0.f,0.f,0.f,0.f};
        #pragma unroll
        for (int ks = 0; ks < 2; ++ks) {
            FragU bu;
            const v4u* p = (const v4u*)(wsfrag + ((4 + nb * 2 + ks) * 32 + l) * 16);
            bu.q[0] = p[0]; bu.q[1] = p[1];
            c = __builtin_amdgcn_wmma_f32_16x16x32_bf16(
                    false, A[ks], false, bu.v, (short)0, c, false, false);
        }
        float bias = b1[nb * 16 + n];
        #pragma unroll
        for (int r = 0; r < 8; ++r) {
            float v = c[r] + bias;
            v = v > 0.f ? v : 0.f;
            lds_h1[w][(r + hb) * HID + nb * 16 + n] = f2bf_raw(v);
        }
    }

    // ---- GEMM2: relu(H1)[16x256] @ W2ᵀ[256x64], epilogue ----
    for (int n2 = 0; n2 < 4; ++n2) {
        v8f c = {0.f,0.f,0.f,0.f,0.f,0.f,0.f,0.f};
        #pragma unroll 2
        for (int ks = 0; ks < 8; ++ks) {
            FragU a;
            const v4u* lp = (const v4u*)&lds_h1[w][m * HID + ks * 32 + hb];
            a.q[0] = lp[0];
            a.q[1] = lp[2];
            FragU bu;
            const v4u* p = (const v4u*)(wsfrag + ((36 + n2 * 8 + ks) * 32 + l) * 16);
            bu.q[0] = p[0]; bu.q[1] = p[1];
            c = __builtin_amdgcn_wmma_f32_16x16x32_bf16(
                    false, a.v, false, bu.v, (short)0, c, false, false);
        }
        int f = n2 * 16 + n;
        float b2f = b2[f], lsf = lscale[f];
        #pragma unroll
        for (int r = 0; r < 8; ++r) {
            int idx = (node0 + r + hb) * FEAT + f;     // lanes 0-15 = 64B run
            out[idx] = lsf * (c[r] + b2f) + x[idx];
        }
    }
}

// ---------------------------------------------------------------------------
extern "C" void kernel_launch(void* const* d_in, const int* in_sizes, int n_in,
                              void* d_out, int out_size, void* d_ws, size_t ws_size,
                              hipStream_t stream) {
    const float* x         = (const float*)d_in[0];
    const float* attr      = (const float*)d_in[1];
    const int*   edge      = (const int*)  d_in[2];
    const float* Wk        = (const float*)d_in[4];
    const float* conv_bias = (const float*)d_in[5];
    const float* gamma     = (const float*)d_in[6];
    const float* beta      = (const float*)d_in[7];
    const float* W1        = (const float*)d_in[8];
    const float* b1        = (const float*)d_in[9];
    const float* W2        = (const float*)d_in[10];
    const float* b2        = (const float*)d_in[11];
    const float* lscale    = (const float*)d_in[12];

    int N = in_sizes[0] / FEAT;
    int E = in_sizes[1] / ATTR;
    const int* src = edge;
    const int* dst = edge + E;

    float* out = (float*)d_out;
    unsigned short* wsfrag = (unsigned short*)d_ws;

    // 1) zero accumulator (d_out doubles as segment-sum buffer)
    int n4 = (N * FEAT) / 4;
    zero_kernel<<<(n4 + 255) / 256, 256, 0, stream>>>((float4*)d_out, n4);

    // 2) pack weights into bf16 WMMA fragments
    prep_weights<<<(68 * 32 + 127) / 128, 128, 0, stream>>>(Wk, W1, W2, wsfrag);

    // 3) edge stage: WMMA depthwise kernel + gather-multiply + atomic scatter
    int tilesE = E / 16;
    int blocksE = 2048;
    int totalWaves = blocksE * (256 / 32);
    edge_kernel<<<blocksE, 256, 0, stream>>>(x, attr, src, dst, wsfrag, out,
                                             tilesE, totalWaves);

    // 4) node stage: LN + WMMA MLP + layer-scale residual
    int tilesN = N / 16;
    node_kernel<<<(tilesN + WPB - 1) / WPB, WPB * 32, 0, stream>>>(
        x, out, conv_bias, gamma, beta, b1, b2, lscale, wsfrag, out, tilesN);
}
